// LstmRnn_34875134443701
// MI455X (gfx1250) — compile-verified, matrix operand order checked by reading, not run on hardware
//
#include <hip/hip_runtime.h>
#include <hip/hip_bf16.h>

typedef __bf16 bf16_t;
typedef __attribute__((ext_vector_type(16))) __bf16 v16bf;
typedef __attribute__((ext_vector_type(8)))  __bf16 v8bf;
typedef __attribute__((ext_vector_type(8)))  float  v8f;

#define BATCH 64
#define TSEQ  512
#define DIN   512
#define HID   1024
#define G4H   4096

#define SCAN_BLOCKS 32
#define GROUP_WAVES 64u                  // waves per batch-tile dependence group
#define LDS_ROW     (HID + 8)            // +16B pad: conflict-free ds_load_b128
#define LDS_TILE    (16 * LDS_ROW)       // elems per 16-row W tile
#define LDS_BYTES   (8 * LDS_TILE * 2)   // 8 tiles per workgroup = 258 KB

// ---------------------------------------------------------------------------
// 16-bit WMMA operand fragment loader (16x32 tile, row-major, ld in elems).
// ISA 7.12.2: lane = row (&15), elems 0..7 = K[kh*8..+7], elems 8..15 =
// K[16+kh*8..+7], kh = lane>>4.  Two 16B loads/lane -> *_load_b128.
// ---------------------------------------------------------------------------
template <typename PTR>
__device__ __forceinline__ v16bf load_frag16(PTR p, int ld) {
    const int lane = threadIdx.x & 31;
    const int row  = lane & 15;
    const int kh   = lane >> 4;
    auto q = p + (size_t)row * ld + kh * 8;
    v8bf lo = *(const v8bf*)(q);
    v8bf hi = *(const v8bf*)(q + 16);
    v16bf r;
#pragma unroll
    for (int i = 0; i < 8; ++i) { r[i] = lo[i]; r[i + 8] = hi[i]; }
    return r;
}

__device__ __forceinline__ float sigmoidf_dev(float x) {
    return 1.0f / (1.0f + __expf(-x));
}

// ---------------------------------------------------------------------------
__global__ void cvt_f32_bf16(const float* __restrict__ in, bf16_t* __restrict__ out, int n) {
    for (int i = blockIdx.x * blockDim.x + threadIdx.x; i < n; i += gridDim.x * blockDim.x)
        out[i] = (bf16_t)in[i];
}

__global__ void zero_ctrs(unsigned* p) {
    if (threadIdx.x < 4 && blockIdx.x == 0) p[threadIdx.x * 64] = 0u;
}

// ---------------------------------------------------------------------------
// xg = x_bf16 (32768 x 512) * W_ih^T + (b_ih + b_hh), fp32 out.
// Wave -> 16x64 tile, k-loop unrolled x2 with explicit ping-pong buffers.
// ---------------------------------------------------------------------------
__global__ void __launch_bounds__(256)
gemm_xg(const bf16_t* __restrict__ xb, const bf16_t* __restrict__ Wih,
        const float* __restrict__ bih, const float* __restrict__ bhh,
        float* __restrict__ xg) {
    const int wave   = blockIdx.x * (blockDim.x >> 5) + (threadIdx.x >> 5);
    const int m_base = (wave >> 6) * 16;
    const int n_base = (wave & 63) * 64;
    const int lane   = threadIdx.x & 31;

    const bf16_t* aptr = xb + (size_t)m_base * DIN;
    const bf16_t* bptr = Wih + (size_t)n_base * DIN;

    v8f acc[4];
#pragma unroll
    for (int j = 0; j < 4; ++j) acc[j] = (v8f)0.0f;

    v16bf a0, a1, b0[4], b1[4];
    a0 = load_frag16(aptr, DIN);
#pragma unroll
    for (int j = 0; j < 4; ++j) b0[j] = load_frag16(bptr + (size_t)(16 * j) * DIN, DIN);

#pragma unroll
    for (int k = 0; k < DIN; k += 64) {
        a1 = load_frag16(aptr + k + 32, DIN);
#pragma unroll
        for (int j = 0; j < 4; ++j)
            b1[j] = load_frag16(bptr + (size_t)(16 * j) * DIN + k + 32, DIN);
#pragma unroll
        for (int j = 0; j < 4; ++j)
            acc[j] = __builtin_amdgcn_wmma_f32_16x16x32_bf16(
                false, a0, false, b0[j], (short)0, acc[j], false, false);
        if (k + 64 < DIN) {
            a0 = load_frag16(aptr + k + 64, DIN);
#pragma unroll
            for (int j = 0; j < 4; ++j)
                b0[j] = load_frag16(bptr + (size_t)(16 * j) * DIN + k + 64, DIN);
        }
#pragma unroll
        for (int j = 0; j < 4; ++j)
            acc[j] = __builtin_amdgcn_wmma_f32_16x16x32_bf16(
                false, a1, false, b1[j], (short)0, acc[j], false, false);
    }

    const int row_h = 8 * (lane >> 4);
    const int col_l = lane & 15;
#pragma unroll
    for (int j = 0; j < 4; ++j) {
        const int n = n_base + 16 * j + col_l;
        const float bias = bih[n] + bhh[n];
#pragma unroll
        for (int r = 0; r < 8; ++r)
            xg[(size_t)(m_base + row_h + r) * G4H + n] = acc[j][r] + bias;
    }
}

// ---------------------------------------------------------------------------
// Per-batch-tile-group quorum: 64 waves arrive (release), spin (acquire).
// Wave32 lockstep reconverges the wave after lane 0's spin; trailing fence
// orders the next step's h loads after the acquire.
// ---------------------------------------------------------------------------
__device__ __forceinline__ void group_sync(unsigned* c, unsigned target) {
    __threadfence();                       // make this wave's stores agent-visible
    if ((threadIdx.x & 31) == 0) {
        __hip_atomic_fetch_add(c, 1u, __ATOMIC_RELEASE, __HIP_MEMORY_SCOPE_AGENT);
        while (__hip_atomic_load(c, __ATOMIC_ACQUIRE, __HIP_MEMORY_SCOPE_AGENT) < target)
            __builtin_amdgcn_s_sleep(1);
    }
    __threadfence();                       // acquire ordering for all lanes
}

// ---------------------------------------------------------------------------
// Persistent LSTM scan: all 512 steps in one launch, 32 blocks x 8 waves.
// Wave w: batch tile bm = w&3, hidden tile bn = blockIdx.x*2 + (w>>2).
// Dependence is confined to each bm group (64 waves): wave reads h rows of
// bm (all cols), writes h rows of bm (its 16 cols) -> per-group sync only.
// W_hh gate tiles staged in LDS once; c / fp32 h live in registers; bf16 h
// ping-pongs through global.
// ---------------------------------------------------------------------------
__global__ void __launch_bounds__(256)
lstm_scan(const bf16_t* __restrict__ Whh, const float* __restrict__ xg,
          const int* __restrict__ mask, const float* __restrict__ h0,
          const float* __restrict__ c0, bf16_t* __restrict__ hping,
          float* __restrict__ hs, unsigned* __restrict__ ctrs) {
    extern __shared__ bf16_t ldsW[];
    const int tid    = threadIdx.x;
    const int wave   = tid >> 5;
    const int lane   = tid & 31;
    const int bm     = wave & 3;          // batch tile 0..3 -> dependence group
    const int bnl    = wave >> 2;         // local hidden tile 0..1
    const int m_base = bm * 16;
    const int n_base = (blockIdx.x * 2 + bnl) * 16;
    unsigned* gctr   = ctrs + bm * 64;    // 256B-separated group counters

    // ---- stage W_hh tiles into LDS (8 tiles: lt = bl*4 + g) ----
    for (int lt = 0; lt < 8; ++lt) {
        const int g  = lt & 3;
        const int gn = (blockIdx.x * 2 + (lt >> 2)) * 16;
        const bf16_t* src = Whh + (size_t)(g * HID + gn) * HID;
        bf16_t* dst = ldsW + lt * LDS_TILE;
        for (int e = tid; e < 16 * (HID / 8); e += 256) {
            const int row = e >> 7, kc = (e & 127) * 8;
            *(v8bf*)(dst + row * LDS_ROW + kc) = *(const v8bf*)(src + (size_t)row * HID + kc);
        }
    }
    const bf16_t* ldsB = ldsW + (bnl * 4) * LDS_TILE;   // this wave's 4 gate tiles

    // ---- init register-resident state + bf16 h buffer 0 ----
    const int n     = n_base + (lane & 15);
    const int row_h = 8 * (lane >> 4);
    v8f hreg, creg;
#pragma unroll
    for (int r = 0; r < 8; ++r) { hreg[r] = h0[n]; creg[r] = c0[n]; }
#pragma unroll
    for (int r = 0; r < 8; ++r)
        hping[(size_t)(m_base + row_h + r) * HID + n] = (bf16_t)hreg[r];

    __syncthreads();                       // LDS weights ready (block-local)
    group_sync(gctr, GROUP_WAVES);         // group's h buffer 0 visible

    const size_t SH = (size_t)BATCH * HID;
    for (int t = 0; t < TSEQ; ++t) {
        const bf16_t* hin  = hping + (size_t)(t & 1) * SH;
        bf16_t*       hout = hping + (size_t)((t & 1) ^ 1) * SH;

        // gates = h_prev * W_hh^T  (A: global bf16 h, ping-pong regs; B: LDS)
        v8f acc[4];
#pragma unroll
        for (int g = 0; g < 4; ++g) acc[g] = (v8f)0.0f;

        const bf16_t* aptr = hin + (size_t)m_base * HID;
        v16bf a0 = load_frag16(aptr, HID);
        v16bf a1;
        for (int k = 0; k < HID; k += 64) {
            a1 = load_frag16(aptr + k + 32, HID);
#pragma unroll
            for (int g = 0; g < 4; ++g) {
                v16bf b = load_frag16(ldsB + g * LDS_TILE + k, LDS_ROW);
                acc[g] = __builtin_amdgcn_wmma_f32_16x16x32_bf16(
                    false, a0, false, b, (short)0, acc[g], false, false);
            }
            if (k + 64 < HID) a0 = load_frag16(aptr + k + 64, HID);
#pragma unroll
            for (int g = 0; g < 4; ++g) {
                v16bf b = load_frag16(ldsB + g * LDS_TILE + k + 32, LDS_ROW);
                acc[g] = __builtin_amdgcn_wmma_f32_16x16x32_bf16(
                    false, a1, false, b, (short)0, acc[g], false, false);
            }
        }

        // fused gate math + mask; c/h carried in registers
#pragma unroll
        for (int r = 0; r < 8; ++r) {
            const int b = m_base + row_h + r;
            const size_t xrow = ((size_t)b * TSEQ + t) * G4H;
            const float gi = acc[0][r] + xg[xrow + 0 * HID + n];
            const float gf = acc[1][r] + xg[xrow + 1 * HID + n];
            const float gg = acc[2][r] + xg[xrow + 2 * HID + n];
            const float go = acc[3][r] + xg[xrow + 3 * HID + n];

            const float ig = sigmoidf_dev(gi);
            const float fg = sigmoidf_dev(gf);
            const float gc = tanhf(gg);
            const float og = sigmoidf_dev(go);

            const float cn = fg * creg[r] + ig * gc;
            const float hn = og * tanhf(cn);

            const bool keep = mask[(size_t)b * TSEQ + t] != 0;
            const float h2 = keep ? hn : hreg[r];
            const float c2 = keep ? cn : creg[r];
            hreg[r] = h2; creg[r] = c2;

            hout[(size_t)b * HID + n] = (bf16_t)h2;
            hs[((size_t)b * TSEQ + t) * HID + n] = h2;
        }

        group_sync(gctr, GROUP_WAVES * (unsigned)(t + 2));  // step boundary
    }
}

// ---------------------------------------------------------------------------
extern "C" void kernel_launch(void* const* d_in, const int* in_sizes, int n_in,
                              void* d_out, int out_size, void* d_ws, size_t ws_size,
                              hipStream_t stream) {
    const float* x    = (const float*)d_in[0];
    const int*   mask = (const int*)  d_in[1];
    const float* Wih  = (const float*)d_in[2];
    const float* Whh  = (const float*)d_in[3];
    const float* bih  = (const float*)d_in[4];
    const float* bhh  = (const float*)d_in[5];
    const float* h0   = (const float*)d_in[6];
    const float* c0   = (const float*)d_in[7];
    float* out = (float*)d_out;

    char* ws = (char*)d_ws;
    float*    xg    = (float*)ws;    ws += (size_t)BATCH * TSEQ * G4H * 4;  // 512 MB
    bf16_t*   hb    = (bf16_t*)ws;   ws += (size_t)2 * BATCH * HID * 2;     // h ping-pong
    bf16_t*   xb    = (bf16_t*)ws;   ws += (size_t)BATCH * TSEQ * DIN * 2;  // x bf16
    bf16_t*   Wih_b = (bf16_t*)ws;   ws += (size_t)G4H * DIN * 2;
    bf16_t*   Whh_b = (bf16_t*)ws;   ws += (size_t)G4H * HID * 2;
    unsigned* ctrs  = (unsigned*)ws; ws += 4 * 256;                         // 4 group ctrs

    zero_ctrs<<<1, 32, 0, stream>>>(ctrs);
    cvt_f32_bf16<<<4096, 256, 0, stream>>>(x,   xb,    BATCH * TSEQ * DIN);
    cvt_f32_bf16<<<2048, 256, 0, stream>>>(Wih, Wih_b, G4H * DIN);
    cvt_f32_bf16<<<2048, 256, 0, stream>>>(Whh, Whh_b, G4H * HID);

    gemm_xg<<<16384, 256, 0, stream>>>(xb, Wih_b, bih, bhh, xg);

    lstm_scan<<<SCAN_BLOCKS, 256, LDS_BYTES, stream>>>(
        Whh_b, xg, mask, h0, c0, hb, out, ctrs);
}
